// DeformableConv2D_85693187489999
// MI455X (gfx1250) — compile-verified, matrix operand order checked by reading, not run on hardware
//
#include <hip/hip_runtime.h>
#include <hip/hip_bf16.h>
#include <cstdint>

// ---------------------------------------------------------------------------
// Deformable Conv2D for MI455X (gfx1250):
//   cols (bilinear-sampled, masked) built in LDS as bf16 B-fragments,
//   weight prepacked as bf16 A-fragments, GEMM via v_wmma_f32_16x16x32_bf16.
//   2x2 register blocking per wave (32F x 32px) -> 4 WMMAs per K-step,
//   halving per-WMMA global and LDS traffic vs 1-tile-per-wave.
// ---------------------------------------------------------------------------

#define Cc 128
#define Hh 64
#define Ww 64
#define Ff 128
#define Bb 8
#define KK 9
#define KTOT (Cc * KK)     // 1152
#define KSTEPS (KTOT / 32) // 36
#define PXT 4              // pixel tiles (of 16) per workgroup
#define PXWG (PXT * 16)    // 64 pixels per workgroup

typedef __attribute__((ext_vector_type(16))) __bf16 v16bf;
typedef __attribute__((ext_vector_type(8)))  float  v8f;

struct alignas(32) Frag8 { uint4 lo, hi; };

__device__ __forceinline__ uint32_t f2bf(float f) {
    uint32_t u = __float_as_uint(f);
    return (u + 0x7FFFu + ((u >> 16) & 1u)) >> 16;       // RNE to bf16
}
__device__ __forceinline__ uint32_t pack2(float a, float b) {
    return f2bf(a) | (f2bf(b) << 16);
}
__device__ __forceinline__ float bfl(uint32_t u) { return __uint_as_float(u << 16); }
__device__ __forceinline__ float bfh(uint32_t u) { return __uint_as_float(u & 0xFFFF0000u); }

// --------------------------- prep 1: NCHW f32 -> NHWC bf16 ------------------
__global__ __launch_bounds__(256) void transpose_x_kernel(
    const float* __restrict__ x, uint32_t* __restrict__ xT /* bf16 pairs */) {
    __shared__ float tile[32][65];
    const int bt  = blockIdx.x;          // 2048 blocks: 512 hw-tiles x 4 c-tiles
    const int c0  = (bt & 3) * 32;
    const int s0  = (bt >> 2) * 64;      // flat (b*H*W); 64 | 4096 -> single b
    const int b   = s0 >> 12;
    const int hw0 = s0 & 4095;
    const int t   = threadIdx.x;
    const int hwl = t & 63;
    const int cl  = t >> 6;              // 0..3
#pragma unroll
    for (int i = 0; i < 8; ++i) {
        int c = cl + i * 4;              // coalesced over hw
        tile[c][hwl] = x[(b * Cc + c0 + c) * 4096 + hw0 + hwl];
    }
    __syncthreads();
    const int dp = t & 15;               // channel pair -> c = 2*dp
    const int hl = t >> 4;               // 0..15
#pragma unroll
    for (int i = 0; i < 4; ++i) {
        int hw = hl + i * 16;
        xT[(s0 + hw) * (Cc / 2) + (c0 >> 1) + dp] =
            pack2(tile[dp * 2][hw], tile[dp * 2 + 1][hw]);
    }
}

// ------------------ prep 2: weight -> bf16 WMMA A-fragment order ------------
// dest dword d = ((ft*36 + ks)*32 + lane)*8 + v ; dword v holds K pair
// kbase = (v&3)*2 + (lane>>4)*8 + (v>>2)*16  (16-bit A 16x32 layout)
__global__ __launch_bounds__(256) void pack_weight_kernel(
    const float* __restrict__ w, uint32_t* __restrict__ wf) {
    int d = blockIdx.x * 256 + threadIdx.x;
    if (d >= (Ff / 16) * KSTEPS * 32 * 8) return;
    int v = d & 7;
    int t = d >> 3;
    int lane = t & 31; t >>= 5;
    int ks = t % KSTEPS;
    int ft = t / KSTEPS;
    int hi = lane >> 4, m = lane & 15;
    int kb = (v & 3) * 2 + hi * 8 + (v >> 2) * 16;
    int f  = ft * 16 + m;
    uint32_t pk = 0;
#pragma unroll
    for (int j = 0; j < 2; ++j) {
        int k = ks * 32 + kb + j;        // col ordering: k = tap*C + c
        int tap = k >> 7, c = k & 127;
        pk |= f2bf(w[(f * Cc + c) * KK + tap]) << (16 * j);
    }
    wf[d] = pk;
}

// ------------------------------ main kernel ---------------------------------
__global__ __launch_bounds__(256) void deform_wmma_kernel(
    const uint4* __restrict__ xT,        // bf16 NHWC, uint4 = 8 channels
    const float* __restrict__ offset,    // (B, 18, H, W)
    const float* __restrict__ mask,      // (B, 9, H, W)
    const uint32_t* __restrict__ wfrag,  // A fragments
    float* __restrict__ out) {           // (B, F, H, W) f32
    // B fragments: [px-tile][ks][lane][8 dwords] = 144 KB (<= 320 KB/WGP)
    __shared__ uint32_t lds[PXT * KSTEPS * 32 * 8];

    const int tid = threadIdx.x;
    // ---- phase 1: build bf16 column B-fragments in LDS ----
    // 64 px x 16 chunks = 1024 units; 4 units per thread
#pragma unroll
    for (int it = 0; it < 4; ++it) {
        const int unit = tid + it * 256;
        const int p   = unit >> 4;                // pixel in group (0..63)
        const int c0  = (unit & 15) * 8;          // 8-channel chunk
        const int pixel = blockIdx.x * PXWG + p;  // flat (b*H*W)
        const int b  = pixel >> 12;
        const int hw = pixel & 4095;
        const int h  = hw >> 6, w = hw & 63;
        const float* offB = offset + b * (2 * KK) * 4096 + hw;
        const float* mskB = mask   + b * KK * 4096 + hw;
        const uint4* xb   = xT + (size_t)b * (Hh * Ww * Cc / 8);
#pragma unroll
        for (int tap = 0; tap < KK; ++tap) {
            const float dy = offB[(2 * tap) * 4096];
            const float dx = offB[(2 * tap + 1) * 4096];
            const float mv = mskB[tap * 4096];
            const int ki = tap / 3, kj = tap % 3;
            const float py  = (float)(h - 1 + ki) + dy;   // pad_t = 1
            const float pxx = (float)(w - 1 + kj) + dx;   // pad_l = 1
            const float y0f = floorf(py), x0f = floorf(pxx);
            const float ly = py - y0f, lx = pxx - x0f;
            const float hy = 1.0f - ly, hx = 1.0f - lx;
            const int y0 = (int)y0f, x0 = (int)x0f;
            const float w00 = hy * hx * mv, w01 = hy * lx * mv;
            const float w10 = ly * hx * mv, w11 = ly * lx * mv;

            uint4 q[4];
            const int ys[2] = { y0, y0 + 1 };
            const int xs[2] = { x0, x0 + 1 };
#pragma unroll
            for (int cy = 0; cy < 2; ++cy)
#pragma unroll
                for (int cx = 0; cx < 2; ++cx) {
                    int yy = ys[cy], xx = xs[cx];
                    uint4 qv = make_uint4(0, 0, 0, 0);
                    if ((unsigned)yy < (unsigned)Hh && (unsigned)xx < (unsigned)Ww)
                        qv = xb[((yy * Ww + xx) * Cc + c0) >> 3];
                    q[cy * 2 + cx] = qv;
                }
            const uint32_t* q00 = &q[0].x; const uint32_t* q01 = &q[1].x;
            const uint32_t* q10 = &q[2].x; const uint32_t* q11 = &q[3].x;

            uint32_t r[4];
#pragma unroll
            for (int i = 0; i < 4; ++i) {
                float lo = w00 * bfl(q00[i]) + w01 * bfl(q01[i]) +
                           w10 * bfl(q10[i]) + w11 * bfl(q11[i]);
                float hi = w00 * bfh(q00[i]) + w01 * bfh(q01[i]) +
                           w10 * bfh(q10[i]) + w11 * bfh(q11[i]);
                r[i] = pack2(lo, hi);
            }
            // B-fragment position for k0 = tap*128 + c0 (4 consecutive dwords)
            const int k0  = tap * Cc + c0;
            const int ks  = k0 >> 5;
            const int kk  = k0 & 31;                  // 0,8,16,24
            const int hi2 = (kk >> 3) & 1;
            const int vb  = (kk >> 4) * 4;
            const int pt  = p >> 4;
            const int ln  = (p & 15) + hi2 * 16;
            *(uint4*)&lds[(((pt * KSTEPS + ks) * 32) + ln) * 8 + vb] =
                make_uint4(r[0], r[1], r[2], r[3]);
        }
    }
    __syncthreads();

    // ---- phase 2: 8 waves, each owns a 2x2 block of 16x16 tiles ----
    const int lane = tid & 31;
    const int wv   = tid >> 5;
    const int ft0  = (wv & 3) * 2;        // F tiles {ft0, ft0+1}
    const int pt0  = (wv >> 2) * 2;       // px tiles {pt0, pt0+1}
    const uint32_t* a0p = wfrag + ((size_t)ft0       * KSTEPS * 32 + lane) * 8;
    const uint32_t* a1p = wfrag + ((size_t)(ft0 + 1) * KSTEPS * 32 + lane) * 8;
    const uint32_t* b0l = &lds[((pt0       * KSTEPS) * 32 + lane) * 8];
    const uint32_t* b1l = &lds[(((pt0 + 1) * KSTEPS) * 32 + lane) * 8];
    v8f acc00 = {}, acc01 = {}, acc10 = {}, acc11 = {};
#pragma unroll 2
    for (int ks = 0; ks < KSTEPS; ++ks) {
        __builtin_prefetch(a0p + (ks + 1) * 256, 0, 1); // global_prefetch next A
        __builtin_prefetch(a1p + (ks + 1) * 256, 0, 1);
        Frag8 a0, a1, b0, b1;
        a0.lo = *(const uint4*)(a0p + ks * 256);
        a0.hi = *(const uint4*)(a0p + ks * 256 + 4);
        a1.lo = *(const uint4*)(a1p + ks * 256);
        a1.hi = *(const uint4*)(a1p + ks * 256 + 4);
        b0.lo = *(const uint4*)(b0l + ks * 256);
        b0.hi = *(const uint4*)(b0l + ks * 256 + 4);
        b1.lo = *(const uint4*)(b1l + ks * 256);
        b1.hi = *(const uint4*)(b1l + ks * 256 + 4);
        v16bf av0 = __builtin_bit_cast(v16bf, a0);
        v16bf av1 = __builtin_bit_cast(v16bf, a1);
        v16bf bv0 = __builtin_bit_cast(v16bf, b0);
        v16bf bv1 = __builtin_bit_cast(v16bf, b1);
        acc00 = __builtin_amdgcn_wmma_f32_16x16x32_bf16(
            false, av0, false, bv0, (short)0, acc00, false, false);
        acc01 = __builtin_amdgcn_wmma_f32_16x16x32_bf16(
            false, av0, false, bv1, (short)0, acc01, false, false);
        acc10 = __builtin_amdgcn_wmma_f32_16x16x32_bf16(
            false, av1, false, bv0, (short)0, acc10, false, false);
        acc11 = __builtin_amdgcn_wmma_f32_16x16x32_bf16(
            false, av1, false, bv1, (short)0, acc11, false, false);
    }
    // ---- store D: VGPR r / lane -> f = 16*ftile + r + 8*(lane>>4),
    //               pixel = base + 16*ptile + (lane&15)
    const int n  = lane & 15, hi = lane >> 4;
    v8f* accs[4] = { &acc00, &acc01, &acc10, &acc11 };
#pragma unroll
    for (int i = 0; i < 2; ++i) {
#pragma unroll
        for (int j = 0; j < 2; ++j) {
            const int pix = blockIdx.x * PXWG + (pt0 + j) * 16 + n;
            const int ob = pix >> 12, ohw = pix & 4095;
            float* op = out + ((size_t)(ob * Ff + (ft0 + i) * 16 + hi * 8)) * 4096 + ohw;
            v8f& a = *accs[i * 2 + j];
#pragma unroll
            for (int r = 0; r < 8; ++r) op[r * 4096] = a[r];
        }
    }
}

// ---------------------------------------------------------------------------
extern "C" void kernel_launch(void* const* d_in, const int* in_sizes, int n_in,
                              void* d_out, int out_size, void* d_ws, size_t ws_size,
                              hipStream_t stream) {
    const float* x      = (const float*)d_in[0];
    const float* offset = (const float*)d_in[1];
    const float* mask   = (const float*)d_in[2];
    const float* weight = (const float*)d_in[3];
    float* out = (float*)d_out;

    // workspace layout: xT (NHWC bf16, 8.39 MB) | weight fragments (0.29 MB)
    uint32_t* xT = (uint32_t*)d_ws;                          // 2,097,152 dwords
    uint32_t* wf = xT + (size_t)Bb * Hh * Ww * (Cc / 2);

    transpose_x_kernel<<<2048, 256, 0, stream>>>(x, xT);
    pack_weight_kernel<<<288, 256, 0, stream>>>(weight, wf);
    deform_wmma_kernel<<<Bb * Hh * Ww / PXWG, 256, 0, stream>>>(
        (const uint4*)xT, offset, mask, wf, out);
}
